// HierarchicalAttention_13494787244224
// MI455X (gfx1250) — matrix operand order repaired and checked by reading, not compiled, not run on hardware
//
#include <hip/hip_runtime.h>

typedef __attribute__((ext_vector_type(16))) _Float16 v16h;
typedef __attribute__((ext_vector_type(8)))  float    v8f;

#define DMODEL 256
#define NHEAD  4
#define DH     64

__device__ __forceinline__ v8f wmma32(v16h a, v16h b, v8f c) {
  return __builtin_amdgcn_wmma_f32_16x16x32_f16(false, a, false, b, (short)0, c, false, false);
}

// A fragment (16x32 f16, M x K) from global f32 row-major; rows >= Lrows -> 0.
// Per-lane layout: m = m0 + lane%16; j<8 -> k = k0 + 8*hi + j ; j>=8 -> k = k0 + 16 + 8*hi + (j-8)
__device__ __forceinline__ v16h afrag_g_f32(const float* base, int stride, int m0, int k0,
                                            int Lrows, int lane) {
  int m = m0 + (lane & 15);
  int hi = lane >> 4;
  v16h a;
  if (m < Lrows) {
    const float* p = base + m * stride + k0 + 8 * hi;
#pragma unroll
    for (int j = 0; j < 8; ++j) a[j] = (_Float16)p[j];
#pragma unroll
    for (int j = 0; j < 8; ++j) a[8 + j] = (_Float16)p[16 + j];
  } else {
#pragma unroll
    for (int j = 0; j < 16; ++j) a[j] = (_Float16)0.f;
  }
  return a;
}

// A fragment from LDS f16 row-major (unguarded / K-guarded variants)
__device__ __forceinline__ v16h afrag_l_f16(const _Float16* base, int stride, int m0, int k0,
                                            int lane) {
  int m = m0 + (lane & 15);
  int hi = lane >> 4;
  const _Float16* p = base + m * stride + k0 + 8 * hi;
  v16h a;
#pragma unroll
  for (int j = 0; j < 8; ++j) a[j] = p[j];
#pragma unroll
  for (int j = 0; j < 8; ++j) a[8 + j] = p[16 + j];
  return a;
}

__device__ __forceinline__ v16h afrag_l_f16g(const _Float16* base, int stride, int m0, int k0,
                                             int kmax, int lane) {
  int m = m0 + (lane & 15);
  int hi = lane >> 4;
  const _Float16* p = base + m * stride + k0 + 8 * hi;
  v16h a;
#pragma unroll
  for (int j = 0; j < 8; ++j) {
    int kk = k0 + 8 * hi + j;
    a[j] = (kk < kmax) ? p[j] : (_Float16)0.f;
  }
#pragma unroll
  for (int j = 0; j < 8; ++j) {
    int kk = k0 + 16 + 8 * hi + j;
    a[8 + j] = (kk < kmax) ? p[16 + j] : (_Float16)0.f;
  }
  return a;
}

// B fragment (32x16 = K x N) from f16 stored transposed Bt[n][k] (k contiguous).
// Per-lane: n = n0 + lane%16 ; j<8 -> k = k0 + 8*hi + j ; j>=8 -> k = k0 + 16 + 8*hi + (j-8)
__device__ __forceinline__ v16h bfrag_t_f16(const _Float16* base, int stride, int n0, int k0,
                                            int lane) {
  int n = n0 + (lane & 15);
  int hi = lane >> 4;
  const _Float16* p = base + n * stride + k0 + 8 * hi;
  v16h b;
#pragma unroll
  for (int j = 0; j < 8; ++j) b[j] = p[j];
#pragma unroll
  for (int j = 0; j < 8; ++j) b[8 + j] = p[16 + j];
  return b;
}

__device__ __forceinline__ v16h bfrag_t_f16g(const _Float16* base, int stride, int n0, int k0,
                                             int kmax, int lane) {
  int n = n0 + (lane & 15);
  int hi = lane >> 4;
  const _Float16* p = base + n * stride + k0 + 8 * hi;
  v16h b;
#pragma unroll
  for (int j = 0; j < 8; ++j) {
    int kk = k0 + 8 * hi + j;
    b[j] = (kk < kmax) ? p[j] : (_Float16)0.f;
  }
#pragma unroll
  for (int j = 0; j < 8; ++j) {
    int kk = k0 + 16 + 8 * hi + j;
    b[8 + j] = (kk < kmax) ? p[16 + j] : (_Float16)0.f;
  }
  return b;
}

// One workgroup (128 threads = 4 wave32) per sequence group.
// Computes masked-mean( MHA(X, valid) ) -> out[g][256] and validout[g] = any(valid).
// Wq/Wk/Wv are pre-converted f16 and pre-transposed to [n][k] (k contiguous) so
// B fragments stream straight from L2 with no LDS staging and no conversion.
// LDS kept at ~38 KB so multiple blocks co-reside per CU to hide VMEM latency.
__global__ __launch_bounds__(128) void mha_pool_kernel(
    const float* __restrict__ X,            // [G, L, 256] f32
    const float* __restrict__ validsrc,     // [G, vstride] (floats, 1/0)
    const _Float16* __restrict__ WqT, const _Float16* __restrict__ WkT,
    const _Float16* __restrict__ WvT,       // [256][256] f16, [n][k]
    const float* __restrict__ Wo,           // [256][256] f32, [k][n]
    const float* __restrict__ bq, const float* __restrict__ bk,
    const float* __restrict__ bv, const float* __restrict__ bo,
    float* __restrict__ out,                // [G, 256]
    float* __restrict__ validout,           // [G] or nullptr
    int L, int Lpad, int vstride) {
  __shared__ _Float16 Qh[64 * 72];       // Q  [token][d]   (head-local)
  __shared__ _Float16 Kh[64 * 72];       // K  [token][d]
  __shared__ _Float16 Vt[64 * 72];       // V^T [d][token]
  __shared__ _Float16 Pp[64 * 72];       // probabilities P [q][k], f16 (A-operand precision)
  __shared__ float    ctxmean[256];      // sum_m w[m] * ctx[m][:]
  __shared__ float    validf_s[64];
  __shared__ float    wvec_s[64];
  __shared__ float    red_s[2];          // V, 1/(V+eps)

  const int tid = threadIdx.x;
  const int lane = tid & 31;
  const int wid = tid >> 5;
  const int hi = lane >> 4;
  const int g = blockIdx.x;
  const int BANDS = Lpad >> 4;           // 16-row bands (1 or 4)
  const bool active = wid < BANDS;
  const int m0 = wid * 16;

  const float* Xg = X + (size_t)g * L * DMODEL;

  for (int i = tid; i < 256; i += 128) ctxmean[i] = 0.f;
  for (int i = tid; i < 64; i += 128)
    validf_s[i] = (i < L) ? validsrc[(size_t)g * vstride + i] : 0.f;
  __syncthreads();
  if (tid == 0) {
    float v = 0.f;
    for (int i = 0; i < 64; ++i) v += validf_s[i];
    float inv = 1.f / (v + 1e-10f);
    red_s[0] = v;
    red_s[1] = inv;
    for (int i = 0; i < 64; ++i) wvec_s[i] = validf_s[i] * inv;
  }
  __syncthreads();

  for (int h = 0; h < NHEAD; ++h) {
    if (active) {
      // ------------- QKV projection for head h (M=Lpad, N=64, K=256) -------------
      v8f qa[4], ka[4], va[4];
#pragma unroll
      for (int nt = 0; nt < 4; ++nt)
#pragma unroll
        for (int r = 0; r < 8; ++r) { qa[nt][r] = 0.f; ka[nt][r] = 0.f; va[nt][r] = 0.f; }

      v16h xa = afrag_g_f32(Xg, DMODEL, m0, 0, L, lane);  // pipelined A fragment
      for (int ks = 0; ks < 8; ++ks) {
        const int k0 = ks * 32;
        const int k0n = (ks == 7) ? k0 : k0 + 32;          // last iter: redundant reload
        v16h xn = afrag_g_f32(Xg, DMODEL, m0, k0n, L, lane);
        // two n-tiles per group: 6 independent loads ahead of 6 WMMAs
#pragma unroll
        for (int p = 0; p < 2; ++p) {
          const int nA = h * DH + (2 * p) * 16;
          const int nB = nA + 16;
          v16h bqA = bfrag_t_f16(WqT, DMODEL, nA, k0, lane);
          v16h bkA = bfrag_t_f16(WkT, DMODEL, nA, k0, lane);
          v16h bvA = bfrag_t_f16(WvT, DMODEL, nA, k0, lane);
          v16h bqB = bfrag_t_f16(WqT, DMODEL, nB, k0, lane);
          v16h bkB = bfrag_t_f16(WkT, DMODEL, nB, k0, lane);
          v16h bvB = bfrag_t_f16(WvT, DMODEL, nB, k0, lane);
          qa[2 * p]     = wmma32(xa, bqA, qa[2 * p]);
          ka[2 * p]     = wmma32(xa, bkA, ka[2 * p]);
          va[2 * p]     = wmma32(xa, bvA, va[2 * p]);
          qa[2 * p + 1] = wmma32(xa, bqB, qa[2 * p + 1]);
          ka[2 * p + 1] = wmma32(xa, bkB, ka[2 * p + 1]);
          va[2 * p + 1] = wmma32(xa, bvB, va[2 * p + 1]);
        }
        xa = xn;
      }
      // write Q/K/V to LDS with bias
#pragma unroll
      for (int nt = 0; nt < 4; ++nt) {
        int n = nt * 16 + (lane & 15);
        float bqv = bq[h * DH + n];
        float bkv = bk[h * DH + n];
        float bvv = bv[h * DH + n];
#pragma unroll
        for (int r = 0; r < 8; ++r) {
          int m = m0 + r + 8 * hi;
          Qh[m * 72 + n] = (_Float16)(qa[nt][r] + bqv);
          Kh[m * 72 + n] = (_Float16)(ka[nt][r] + bkv);
          Vt[n * 72 + m] = (_Float16)(va[nt][r] + bvv);  // transposed: [d][token]
        }
      }
    }
    __syncthreads();  // K/V visible to all waves

    // ---------------- attention for this wave's 16-query band ----------------
    if (active) {
      v8f sacc[4];
#pragma unroll
      for (int nt = 0; nt < 4; ++nt)
#pragma unroll
        for (int r = 0; r < 8; ++r) sacc[nt][r] = 0.f;

#pragma unroll
      for (int kd = 0; kd < 2; ++kd) {  // dh = 64 -> 2 K-steps
        v16h qf = afrag_l_f16(Qh, 72, m0, kd * 32, lane);
        v16h kf[4];
#pragma unroll
        for (int nt = 0; nt < 4; ++nt)
          if (nt < BANDS) kf[nt] = bfrag_t_f16(Kh, 72, nt * 16, kd * 32, lane);
#pragma unroll
        for (int nt = 0; nt < 4; ++nt)
          if (nt < BANDS) sacc[nt] = wmma32(qf, kf[nt], sacc[nt]);
      }

      // scale + key mask + softmax (rows split: m = r + 8*hi across 16-lane halves)
      float vkey[4];
#pragma unroll
      for (int nt = 0; nt < 4; ++nt)
        vkey[nt] = (nt < BANDS) ? validf_s[nt * 16 + (lane & 15)] : 0.f;
#pragma unroll
      for (int nt = 0; nt < 4; ++nt)
#pragma unroll
        for (int r = 0; r < 8; ++r) {
          float s = sacc[nt][r] * 0.125f;  // 1/sqrt(64)
          sacc[nt][r] = (vkey[nt] > 0.f) ? s : -1e9f;
        }
#pragma unroll
      for (int r = 0; r < 8; ++r) {
        float mx = -3.0e38f;
#pragma unroll
        for (int nt = 0; nt < 4; ++nt)
          if (nt < BANDS) mx = fmaxf(mx, sacc[nt][r]);
#pragma unroll
        for (int off = 1; off < 16; off <<= 1) mx = fmaxf(mx, __shfl_xor(mx, off, 16));
        float sum = 0.f;
#pragma unroll
        for (int nt = 0; nt < 4; ++nt)
          if (nt < BANDS) {
            float e = __expf(sacc[nt][r] - mx);
            sacc[nt][r] = e;
            sum += e;
          }
#pragma unroll
        for (int off = 1; off < 16; off <<= 1) sum += __shfl_xor(sum, off, 16);
        float inv = 1.f / sum;
#pragma unroll
        for (int nt = 0; nt < 4; ++nt)
          if (nt < BANDS) sacc[nt][r] *= inv;
      }
      // park P in LDS as f16 (own rows only; in-wave DS ordering suffices)
#pragma unroll
      for (int nt = 0; nt < 4; ++nt)
        if (nt < BANDS)
#pragma unroll
          for (int r = 0; r < 8; ++r)
            Pp[(m0 + r + 8 * hi) * 72 + nt * 16 + (lane & 15)] = (_Float16)sacc[nt][r];

      // ctx = P @ V  (M=16, N=64, K=Lpad)
      v8f cacc[4];
#pragma unroll
      for (int nt = 0; nt < 4; ++nt)
#pragma unroll
        for (int r = 0; r < 8; ++r) cacc[nt][r] = 0.f;
      if (Lpad == 64) {
        // fast path: no K-guards; group loads ahead of WMMAs
        for (int kt = 0; kt < 2; ++kt) {
          int k0 = kt * 32;
          v16h pf = afrag_l_f16(Pp, 72, m0, k0, lane);
          v16h vf[4];
#pragma unroll
          for (int nt = 0; nt < 4; ++nt) vf[nt] = bfrag_t_f16(Vt, 72, nt * 16, k0, lane);
#pragma unroll
          for (int nt = 0; nt < 4; ++nt) cacc[nt] = wmma32(pf, vf[nt], cacc[nt]);
        }
      } else {
        // Lpad == 16: single K-step with zero-guard on the 16..31 tail
        v16h pf = afrag_l_f16g(Pp, 72, m0, 0, Lpad, lane);
        v16h vf[4];
#pragma unroll
        for (int nt = 0; nt < 4; ++nt) vf[nt] = bfrag_t_f16g(Vt, 72, nt * 16, 0, Lpad, lane);
#pragma unroll
        for (int nt = 0; nt < 4; ++nt) cacc[nt] = wmma32(pf, vf[nt], cacc[nt]);
      }
      // weighted pooling of context (masked-mean moved before the Wo projection)
#pragma unroll
      for (int nt = 0; nt < 4; ++nt) {
        float acc = 0.f;
#pragma unroll
        for (int r = 0; r < 8; ++r) acc += wvec_s[m0 + r + 8 * hi] * cacc[nt][r];
        acc += __shfl_xor(acc, 16, 32);
        if (lane < 16) atomicAdd(&ctxmean[h * DH + nt * 16 + lane], acc);
      }
    }
    __syncthreads();  // ctxmean complete; also protects Qh/Kh/Vt reuse next head
  }

  // pooled = ctxmean @ Wo + bo * V/(V+eps)   (tiny GEMV; weights stay hot in L2)
  float ratio = red_s[0] * red_s[1];
  for (int d = tid; d < 256; d += 128) {
    float acc = 0.f;
    for (int k2 = 0; k2 < 256; ++k2) acc = fmaf(ctxmean[k2], Wo[(size_t)k2 * DMODEL + d], acc);
    out[(size_t)g * DMODEL + d] = acc + bo[d] * ratio;
  }
  if (validout && tid == 0) validout[g] = (red_s[0] > 0.f) ? 1.f : 0.f;
}

__global__ void mask_to_float_kernel(const int* __restrict__ m, float* __restrict__ o, int n) {
  int i = blockIdx.x * blockDim.x + threadIdx.x;
  if (i < n) o[i] = (float)m[i];
}

// WT[n][k] = (f16) W[k][n]  for one 256x256 matrix
__global__ void w_transpose_f16_kernel(const float* __restrict__ W, _Float16* __restrict__ WT) {
  int idx = blockIdx.x * blockDim.x + threadIdx.x;  // 65536
  int n = idx >> 8;
  int k = idx & 255;
  WT[idx] = (_Float16)W[(size_t)k * DMODEL + n];
}

extern "C" void kernel_launch(void* const* d_in, const int* in_sizes, int n_in,
                              void* d_out, int out_size, void* d_ws, size_t ws_size,
                              hipStream_t stream) {
  (void)in_sizes; (void)n_in; (void)out_size; (void)ws_size;
  const float* wemb  = (const float*)d_in[0];
  const int*   amask = (const int*)d_in[1];
  const float* wWq = (const float*)d_in[2],  *wWk = (const float*)d_in[3];
  const float* wWv = (const float*)d_in[4],  *wWo = (const float*)d_in[5];
  const float* wbq = (const float*)d_in[6],  *wbk = (const float*)d_in[7];
  const float* wbv = (const float*)d_in[8],  *wbo = (const float*)d_in[9];
  const float* sWq = (const float*)d_in[10], *sWk = (const float*)d_in[11];
  const float* sWv = (const float*)d_in[12], *sWo = (const float*)d_in[13];
  const float* sbq = (const float*)d_in[14], *sbk = (const float*)d_in[15];
  const float* sbv = (const float*)d_in[16], *sbo = (const float*)d_in[17];
  const float* cWq = (const float*)d_in[18], *cWk = (const float*)d_in[19];
  const float* cWv = (const float*)d_in[20], *cWo = (const float*)d_in[21];
  const float* cbq = (const float*)d_in[22], *cbk = (const float*)d_in[23];
  const float* cbv = (const float*)d_in[24], *cbo = (const float*)d_in[25];

  // workspace layout
  float* ws     = (float*)d_ws;
  float* validw = ws;                   // [2048*64]
  float* sent   = validw + 2048 * 64;   // [2048*256]
  float* smask  = sent + 2048 * 256;    // [2048]   = [128 groups][16]
  float* sec    = smask + 2048;         // [128*256]
  float* cmask  = sec + 128 * 256;      // [128]    = [16 groups][8]
  _Float16* wt  = (_Float16*)(cmask + 128);  // 9 x 65536 halves, f16 transposed weights
  _Float16* wqT = wt + 0 * 65536; _Float16* wkT = wt + 1 * 65536; _Float16* wvT = wt + 2 * 65536;
  _Float16* sqT = wt + 3 * 65536; _Float16* skT = wt + 4 * 65536; _Float16* svT = wt + 5 * 65536;
  _Float16* cqT = wt + 6 * 65536; _Float16* ckT = wt + 7 * 65536; _Float16* cvT = wt + 8 * 65536;

  const int nmask = 16 * 8 * 16 * 64;
  mask_to_float_kernel<<<(nmask + 255) / 256, 256, 0, stream>>>(amask, validw, nmask);

  // pre-transpose + convert the Q/K/V weights of all three levels to f16 [n][k]
  w_transpose_f16_kernel<<<256, 256, 0, stream>>>(wWq, wqT);
  w_transpose_f16_kernel<<<256, 256, 0, stream>>>(wWk, wkT);
  w_transpose_f16_kernel<<<256, 256, 0, stream>>>(wWv, wvT);
  w_transpose_f16_kernel<<<256, 256, 0, stream>>>(sWq, sqT);
  w_transpose_f16_kernel<<<256, 256, 0, stream>>>(sWk, skT);
  w_transpose_f16_kernel<<<256, 256, 0, stream>>>(sWv, svT);
  w_transpose_f16_kernel<<<256, 256, 0, stream>>>(cWq, cqT);
  w_transpose_f16_kernel<<<256, 256, 0, stream>>>(cWk, ckT);
  w_transpose_f16_kernel<<<256, 256, 0, stream>>>(cWv, cvT);

  // word level: 2048 sentences of 64 tokens
  mha_pool_kernel<<<2048, 128, 0, stream>>>(wemb, validw, wqT, wkT, wvT, wWo,
                                            wbq, wbk, wbv, wbo, sent, smask,
                                            64, 64, 64);
  // sentence level: 128 sections of 16 sentences
  mha_pool_kernel<<<128, 128, 0, stream>>>(sent, smask, sqT, skT, svT, sWo,
                                           sbq, sbk, sbv, sbo, sec, cmask,
                                           16, 16, 16);
  // section level: 16 docs of 8 sections (padded to 16)
  mha_pool_kernel<<<16, 128, 0, stream>>>(sec, cmask, cqT, ckT, cvT, cWo,
                                          cbq, cbk, cbv, cbo, (float*)d_out, nullptr,
                                          8, 16, 8);
}
